// DCTFrequencyFilterGVA_68985764708924
// MI455X (gfx1250) — matrix operand order repaired and checked by reading, not compiled, not run on hardware
//
#include <hip/hip_runtime.h>

// ---------------------------------------------------------------------------
// DCTFrequencyFilterGVA for MI455X (gfx1250, wave32, WMMA f16 16x16x32)
// ---------------------------------------------------------------------------

#define N_PTS 65536
#define S_NB  16
#define C_CH  256
#define G_GR  16
#define EPS_BN 1e-5f
#define PI_F 3.14159265358979323846f

typedef __attribute__((ext_vector_type(16))) _Float16 v16h;
typedef __attribute__((ext_vector_type(8)))  float    v8f;

__device__ __forceinline__ float reluf(float x) { return x > 0.f ? x : 0.f; }

// ------------------------- setup / swizzle kernels -------------------------

// [Wq|Wk|Wv] (256 x 768) -> f16 WMMA-B fragment layout:
// e = ((kc*48+nt)*32 + lane)*16 + j ; K = kc*32 + (lane>=16?8:0) + (j&7) + (j>=8?16:0),
// col = nt*16 + (lane&15)
__global__ __launch_bounds__(256) void k_swz_wall(const float* __restrict__ Wq,
                                                  const float* __restrict__ Wk,
                                                  const float* __restrict__ Wv,
                                                  _Float16* __restrict__ outh) {
  int e = blockIdx.x * 256 + threadIdx.x;        // < 8*48*32*16 = 196608
  int j = e & 15;
  int lane = (e >> 4) & 31;
  int rest = e >> 9;                             // kc*48 + nt
  int kc = rest / 48;
  int nt = rest % 48;
  int k = kc * 32 + ((lane >= 16) ? 8 : 0) + (j & 7) + ((j >= 8) ? 16 : 0);
  int col = nt * 16 + (lane & 15);
  float v;
  if (col < 256)      v = Wq[k * 256 + col];
  else if (col < 512) v = Wk[k * 256 + col - 256];
  else                v = Wv[k * 256 + col - 512];
  outh[e] = (_Float16)v;
}

// Wp2 (256 x 256) -> f16 fragment layout, e = ((kc*16+nt)*32+lane)*16+j
__global__ __launch_bounds__(256) void k_swz_wp2(const float* __restrict__ Wp2,
                                                 _Float16* __restrict__ outh) {
  int e = blockIdx.x * 256 + threadIdx.x;        // < 65536
  int j = e & 15;
  int lane = (e >> 4) & 31;
  int rest = e >> 9;                             // kc*16 + nt
  int kc = rest >> 4;
  int nt = rest & 15;
  int k = kc * 32 + ((lane >= 16) ? 8 : 0) + (j & 7) + ((j >= 8) ? 16 : 0);
  int col = nt * 16 + (lane & 15);
  outh[e] = (_Float16)Wp2[k * 256 + col];
}

// Wp2g[k][g] = sum_i Wp2[k][16g+i]*glw[16g+i], emitted directly in f16
// B-fragment layout (single N-tile): e = (kc*32+lane)*16+j
__global__ __launch_bounds__(256) void k_swz_wp2g(const float* __restrict__ Wp2,
                                                  const float* __restrict__ glw,
                                                  _Float16* __restrict__ outh) {
  int e = blockIdx.x * 256 + threadIdx.x;        // < 8*32*16 = 4096
  int j = e & 15;
  int lane = (e >> 4) & 31;
  int kc = e >> 9;
  int k = kc * 32 + ((lane >= 16) ? 8 : 0) + (j & 7) + ((j >= 8) ? 16 : 0);
  int g = lane & 15;
  float a = 0.f;
  for (int i = 0; i < 16; ++i) a += Wp2[k * 256 + g * 16 + i] * glw[g * 16 + i];
  outh[e] = (_Float16)a;
}

__device__ __forceinline__ float dct_D(int j, int m) {
  if (j == 0) return 0.25f;  // sqrt(1/16)
  return sqrtf(2.f / 16.f) * cosf(PI_F * (float)j * (float)(2 * m + 1) / 32.f);
}
__device__ __forceinline__ float freq_F(int j) {
  if (j < 4) return 2.0f;
  if (j < 11) {
    float cw = 0.5f * (1.f + cosf(PI_F * (float)(j - 4) / 6.f));
    return 1.f + cw;  // 2*cw + 1*(1-cw)
  }
  float cw = 0.5f * (1.f + cosf(PI_F * (float)(j - 11) / 4.f));
  return 0.1f + 0.9f * cw;
}

// bpg[g] = sum_i bp2*glw ; E[m][i] = sum_j D[j,m]*F[j]*D[j,i]
__global__ __launch_bounds__(256) void k_prep_misc(const float* __restrict__ bp2,
                                                   const float* __restrict__ glw,
                                                   float* __restrict__ bpg,
                                                   float* __restrict__ Emat) {
  int t = threadIdx.x;
  if (t < 16) {
    float a = 0.f;
    for (int i = 0; i < 16; ++i) a += bp2[t * 16 + i] * glw[t * 16 + i];
    bpg[t] = a;
  }
  int m = t >> 4, i2 = t & 15;
  float acc = 0.f;
  for (int j = 0; j < 16; ++j) acc += dct_D(j, m) * freq_F(j) * dct_D(j, i2);
  Emat[t] = acc;
}

// ----------------------------- QKV GEMM (WMMA) -----------------------------
// One wave per 16-row tile: out (16 x 768) = feat_tile (16x256) @ [Wq|Wk|Wv]
__global__ __launch_bounds__(256) void k_qkv(const float* __restrict__ feat,
                                             const _Float16* __restrict__ Wsw,
                                             const float* __restrict__ bv,
                                             float* __restrict__ qlin,
                                             float* __restrict__ klin,
                                             float* __restrict__ vout) {
  int wave = blockIdx.x * 8 + (threadIdx.x >> 5);
  int lane = threadIdx.x & 31;
  int hi = lane >> 4, li = lane & 15;
  int row0 = wave * 16;

  // A fragments: two contiguous 8-float runs per k-chunk -> float4 loads
  v16h a[8];
  const float4* f4row = (const float4*)(feat + (size_t)(row0 + li) * C_CH);
#pragma unroll
  for (int kc = 0; kc < 8; ++kc) {
    int kb4 = (kc * 32 + (hi ? 8 : 0)) >> 2;
    float4 a0 = f4row[kb4 + 0];
    float4 a1 = f4row[kb4 + 1];
    float4 a2 = f4row[kb4 + 4];
    float4 a3 = f4row[kb4 + 5];
    a[kc][0]  = (_Float16)a0.x; a[kc][1]  = (_Float16)a0.y;
    a[kc][2]  = (_Float16)a0.z; a[kc][3]  = (_Float16)a0.w;
    a[kc][4]  = (_Float16)a1.x; a[kc][5]  = (_Float16)a1.y;
    a[kc][6]  = (_Float16)a1.z; a[kc][7]  = (_Float16)a1.w;
    a[kc][8]  = (_Float16)a2.x; a[kc][9]  = (_Float16)a2.y;
    a[kc][10] = (_Float16)a2.z; a[kc][11] = (_Float16)a2.w;
    a[kc][12] = (_Float16)a3.x; a[kc][13] = (_Float16)a3.y;
    a[kc][14] = (_Float16)a3.z; a[kc][15] = (_Float16)a3.w;
  }

  const v16h* Bf = (const v16h*)Wsw;
  for (int nt = 0; nt < 48; ++nt) {
    // batch all 8 B-fragment loads (one clause, one wait), then 8 WMMAs
    v16h b[8];
#pragma unroll
    for (int kc = 0; kc < 8; ++kc) b[kc] = Bf[(kc * 48 + nt) * 32 + lane];
    v8f acc = {};
#pragma unroll
    for (int kc = 0; kc < 8; ++kc)
      acc = __builtin_amdgcn_wmma_f32_16x16x32_f16(false, a[kc], false, b[kc],
                                                   (short)0, acc, false, false);
    int col = nt * 16 + li;
    float* dst;
    int cc;
    float bias = 0.f;
    if (col < 256)      { dst = qlin; cc = col; }          // bq cancels in BN
    else if (col < 512) { dst = klin; cc = col - 256; }    // bk cancels in BN
    else                { dst = vout; cc = col - 512; bias = bv[cc]; }
#pragma unroll
    for (int r = 0; r < 8; ++r) {
      int m = r + (hi ? 8 : 0);
      dst[(size_t)(row0 + m) * C_CH + cc] = acc[r] + bias;
    }
  }
}

// --------------------- column stats (q_lin / k_lin BN) ---------------------
__global__ __launch_bounds__(256) void k_colstats(const float* __restrict__ qlin,
                                                  const float* __restrict__ klin,
                                                  float* __restrict__ part) {
  int b = blockIdx.x, c = threadIdx.x;  // 256 blocks, 256 rows each
  float qs = 0, qss = 0, ks = 0, kss = 0;
  for (int r = 0; r < 256; ++r) {
    size_t off = (size_t)(b * 256 + r) * C_CH + c;
    float q = qlin[off]; qs += q; qss += q * q;
    float k = klin[off]; ks += k; kss += k * k;
  }
  part[(0 * 256 + b) * 256 + c] = qs;
  part[(1 * 256 + b) * 256 + c] = qss;
  part[(2 * 256 + b) * 256 + c] = ks;
  part[(3 * 256 + b) * 256 + c] = kss;
}

__global__ __launch_bounds__(256) void k_qk_final(const float* __restrict__ part,
                                                  const float* __restrict__ gq,
                                                  const float* __restrict__ betaq,
                                                  const float* __restrict__ gk,
                                                  const float* __restrict__ betak,
                                                  float* __restrict__ scq, float* __restrict__ shq,
                                                  float* __restrict__ sck, float* __restrict__ shk) {
  int c = threadIdx.x;
  float qs = 0, qss = 0, ks = 0, kss = 0;
  for (int b = 0; b < 256; ++b) {
    qs += part[(0 * 256 + b) * 256 + c];
    qss += part[(1 * 256 + b) * 256 + c];
    ks += part[(2 * 256 + b) * 256 + c];
    kss += part[(3 * 256 + b) * 256 + c];
  }
  float inv = 1.f / (float)N_PTS;
  float mq = qs * inv, vq = qss * inv - mq * mq;
  float mk = ks * inv, vk = kss * inv - mk * mk;
  float sq = gq[c] * rsqrtf(vq + EPS_BN);
  float sk = gk[c] * rsqrtf(vk + EPS_BN);
  scq[c] = sq; shq[c] = betaq[c] - mq * sq;
  sck[c] = sk; shk[c] = betak[c] - mk * sk;
}

// qg[n,g] = sum_i relu(bn(qlin))*glw ; kg likewise
__global__ __launch_bounds__(256) void k_qg_kg(const float* __restrict__ qlin,
                                               const float* __restrict__ klin,
                                               const float* __restrict__ scq, const float* __restrict__ shq,
                                               const float* __restrict__ sck, const float* __restrict__ shk,
                                               const float* __restrict__ glw,
                                               float* __restrict__ qg, float* __restrict__ kg) {
  int t = blockIdx.x * 256 + threadIdx.x;  // < N*G
  int n = t >> 4, g = t & 15;
  float qa = 0.f, ka = 0.f;
  for (int i = 0; i < 16; ++i) {
    int c = g * 16 + i;
    float qv = reluf(scq[c] * qlin[(size_t)n * C_CH + c] + shq[c]);
    float kv = reluf(sck[c] * klin[(size_t)n * C_CH + c] + shk[c]);
    qa += qv * glw[c];
    ka += kv * glw[c];
  }
  qg[t] = qa; kg[t] = ka;
}

// ---------------------- pos-linear BN stats (axes 0,1) ---------------------
__global__ __launch_bounds__(256) void k_pos_stats(const float* __restrict__ coord,
                                                   const int* __restrict__ ref,
                                                   const float* __restrict__ Wp1,
                                                   const float* __restrict__ bp1,
                                                   float* __restrict__ part) {
  __shared__ float pp[256][3];
  int b = blockIdx.x, t = threadIdx.x;
  float w1a = Wp1[t], w1b = Wp1[C_CH + t], w1c = Wp1[2 * C_CH + t], bb = bp1[t];
  float s = 0.f, ss = 0.f;
  for (int ch = 0; ch < 16; ++ch) {
    int p = b * 4096 + ch * 256 + t;  // < N*S
    int n = p >> 4;
    int ii = ref[p]; if (ii < 0) ii = 0;
    pp[t][0] = coord[ii * 3 + 0] - coord[n * 3 + 0];
    pp[t][1] = coord[ii * 3 + 1] - coord[n * 3 + 1];
    pp[t][2] = coord[ii * 3 + 2] - coord[n * 3 + 2];
    __syncthreads();
    for (int q2 = 0; q2 < 256; ++q2) {
      float pl = pp[q2][0] * w1a + pp[q2][1] * w1b + pp[q2][2] * w1c + bb;
      s += pl; ss += pl * pl;
    }
    __syncthreads();
  }
  part[b * 256 + t] = s;
  part[65536 + b * 256 + t] = ss;
}

// finalize pos BN and pack per-channel params: ppk[c] = {w1x,w1y,w1z,bp1,scp,shp,0,0}
__global__ __launch_bounds__(256) void k_pos_final(const float* __restrict__ part,
                                                   const float* __restrict__ gp,
                                                   const float* __restrict__ betap,
                                                   const float* __restrict__ Wp1,
                                                   const float* __restrict__ bp1,
                                                   float* __restrict__ ppk) {
  int c = threadIdx.x;
  float s = 0.f, ss = 0.f;
  for (int b = 0; b < 256; ++b) { s += part[b * 256 + c]; ss += part[65536 + b * 256 + c]; }
  float inv = 1.f / (float)(N_PTS * S_NB);
  float m = s * inv, v = ss * inv - m * m;
  float sc = gp[c] * rsqrtf(v + EPS_BN);
  float sh = betap[c] - m * sc;
  ppk[c * 8 + 0] = Wp1[c];
  ppk[c * 8 + 1] = Wp1[C_CH + c];
  ppk[c * 8 + 2] = Wp1[2 * C_CH + c];
  ppk[c * 8 + 3] = bp1[c];
  ppk[c * 8 + 4] = sc;
  ppk[c * 8 + 5] = sh;
  ppk[c * 8 + 6] = 0.f;
  ppk[c * 8 + 7] = 0.f;
}

// shared: build A fragments (h rows) for point n, lane-local, from packed params
__device__ __forceinline__ void build_a_frags(const float* __restrict__ ppk,
                                              float px, float py, float pz,
                                              int hi, v16h a[8]) {
#pragma unroll
  for (int kc = 0; kc < 8; ++kc) {
#pragma unroll
    for (int j = 0; j < 16; ++j) {
      int k = kc * 32 + (hi ? 8 : 0) + (j & 7) + ((j >= 8) ? 16 : 0);
      const float4* P = (const float4*)(ppk + k * 8);
      float4 p0 = P[0];
      float4 p1 = P[1];
      float pl = px * p0.x + py * p0.y + pz * p0.z + p0.w;
      a[kc][j] = (_Float16)reluf(p1.x * pl + p1.y);
    }
  }
}

// ------------- w0[n,s,g] = kg[idx]-qg + h@Wp2g + bpg  (WMMA, 1 wave/point) --
__global__ __launch_bounds__(256) void k_w0(const float* __restrict__ coord,
                                            const int* __restrict__ ref,
                                            const float* __restrict__ ppk,
                                            const _Float16* __restrict__ wp2gsw,
                                            const float* __restrict__ bpg,
                                            const float* __restrict__ qg,
                                            const float* __restrict__ kg,
                                            float* __restrict__ w0) {
  int wv = blockIdx.x * 8 + (threadIdx.x >> 5);
  int lane = threadIdx.x & 31;
  int hi = lane >> 4, li = lane & 15;
  int n = wv;

  int ir = ref[n * 16 + li];
  int ic = ir < 0 ? 0 : ir;
  float px = coord[ic * 3 + 0] - coord[n * 3 + 0];
  float py = coord[ic * 3 + 1] - coord[n * 3 + 1];
  float pz = coord[ic * 3 + 2] - coord[n * 3 + 2];

  v16h a[8];
  build_a_frags(ppk, px, py, pz, hi, a);

  v16h b[8];
  const v16h* Bf = (const v16h*)wp2gsw;
#pragma unroll
  for (int kc = 0; kc < 8; ++kc) b[kc] = Bf[kc * 32 + lane];
  v8f acc = {};
#pragma unroll
  for (int kc = 0; kc < 8; ++kc)
    acc = __builtin_amdgcn_wmma_f32_16x16x32_f16(false, a[kc], false, b[kc],
                                                 (short)0, acc, false, false);

  float qv = qg[n * 16 + li];
  float bv0 = bpg[li];
#pragma unroll
  for (int r = 0; r < 8; ++r) {
    int m = r + (hi ? 8 : 0);
    int ii = ref[n * 16 + m];
    int ic2 = ii < 0 ? 0 : ii;
    w0[((size_t)n * 16 + m) * 16 + li] = acc[r] + kg[ic2 * 16 + li] - qv + bv0;
  }
}

// ----------------------- w0 group stats (axes 0,1) -------------------------
__global__ __launch_bounds__(256) void k_w0stats(const float* __restrict__ w0,
                                                 float* __restrict__ wpart) {
  __shared__ float sh[256], sh2[256];
  int b = blockIdx.x, t = threadIdx.x;
  int g = t & 15, sub = t >> 4;
  float s = 0.f, ss = 0.f;
  for (int i = 0; i < 256; ++i) {
    int p = b * 4096 + i * 16 + sub;
    float v = w0[(size_t)p * 16 + g];
    s += v; ss += v * v;
  }
  sh[t] = s; sh2[t] = ss;
  __syncthreads();
  if (t < 16) {
    float a = 0.f, a2 = 0.f;
    for (int k = 0; k < 16; ++k) { a += sh[k * 16 + t]; a2 += sh2[k * 16 + t]; }
    wpart[b * 16 + t] = a;
    wpart[4096 + b * 16 + t] = a2;
  }
}

__global__ __launch_bounds__(256) void k_w_final(const float* __restrict__ wpart,
                                                 const float* __restrict__ gw,
                                                 const float* __restrict__ betaw,
                                                 float* __restrict__ wsc, float* __restrict__ wsh) {
  int g = threadIdx.x;
  if (g >= 16) return;
  float s = 0.f, ss = 0.f;
  for (int b = 0; b < 256; ++b) { s += wpart[b * 16 + g]; ss += wpart[4096 + b * 16 + g]; }
  float inv = 1.f / (float)(N_PTS * S_NB);
  float m = s * inv, v = ss * inv - m * m;
  float sc = gw[g] * rsqrtf(v + EPS_BN);
  wsc[g] = sc; wsh[g] = betaw[g] - m * sc;
}

// -------------------------------- main fused -------------------------------
// One wave per point. Wp2 (f16, fragment layout) staged in LDS once/block.
__global__ __launch_bounds__(256) void k_main(
    const float* __restrict__ coord, const int* __restrict__ ref,
    const float* __restrict__ vall, const float* __restrict__ w0,
    const _Float16* __restrict__ wp2sw,
    const float* __restrict__ ppk,
    const float* __restrict__ bp2,
    const float* __restrict__ wsc, const float* __restrict__ wsh,
    const float* __restrict__ Ww, const float* __restrict__ bw,
    const float* __restrict__ Emat,
    const float* __restrict__ Wg1, const float* __restrict__ bg1,
    const float* __restrict__ Wg2, const float* __restrict__ bg2,
    float* __restrict__ out) {
  __shared__ __align__(32) _Float16 wp2s[C_CH * C_CH];  // 128 KB
  __shared__ float wbuf[8][16][17];                     // per-wave w tiles

  int tid = threadIdx.x;
  int w = tid >> 5, lane = tid & 31, hi = lane >> 4, li = lane & 15;
  int n = blockIdx.x * 8 + w;

  // stage Wp2 fragments into LDS
  {
    const uint4* src = (const uint4*)wp2sw;
    uint4* dst = (uint4*)wp2s;
    for (int i = tid; i < (C_CH * C_CH * 2) / 16; i += 256) dst[i] = src[i];
  }
  __syncthreads();

  // ---- neighbor-weight softmax: BN -> relu -> @Ww -> softmax(s) -> mask ----
  if (lane < 16) {
#pragma unroll
    for (int s = 0; s < 16; ++s) {
      float v0 = w0[((size_t)n * 16 + s) * 16 + li];
      wbuf[w][s][li] = reluf(wsc[li] * v0 + wsh[li]);
    }
  }
  __syncthreads();
  float w2r[16];
  if (lane < 16) {
#pragma unroll
    for (int s = 0; s < 16; ++s) {
      float a2 = bw[li];
#pragma unroll
      for (int g = 0; g < 16; ++g) a2 += wbuf[w][s][g] * Ww[g * 16 + li];
      w2r[s] = a2;
    }
  }
  __syncthreads();
  if (lane < 16) {
    float mx = w2r[0];
#pragma unroll
    for (int s = 1; s < 16; ++s) mx = fmaxf(mx, w2r[s]);
    float sum = 0.f;
#pragma unroll
    for (int s = 0; s < 16; ++s) { float e = __expf(w2r[s] - mx); w2r[s] = e; sum += e; }
    float inv = 1.f / sum;
#pragma unroll
    for (int s = 0; s < 16; ++s) {
      float msk = (ref[n * 16 + s] >= 0) ? 1.f : 0.f;
      wbuf[w][s][li] = w2r[s] * inv * msk;
    }
  }
  __syncthreads();

  // ---- A fragments: h[s=M, k] = relu(bn(pos_M @ Wp1 + bp1)), f16 ----------
  int ir = ref[n * 16 + li];
  int ic = ir < 0 ? 0 : ir;
  float px = coord[ic * 3 + 0] - coord[n * 3 + 0];
  float py = coord[ic * 3 + 1] - coord[n * 3 + 1];
  float pz = coord[ic * 3 + 2] - coord[n * 3 + 2];

  v16h a[8];
  build_a_frags(ppk, px, py, pz, hi, a);

  int idxv[8];
#pragma unroll
  for (int r = 0; r < 8; ++r) {
    int m = r + (hi ? 8 : 0);
    int ii = ref[n * 16 + m];
    idxv[r] = ii < 0 ? 0 : ii;
  }

  // ---- per-group: peb tile via WMMA, fuse value add + weighted s-sum ------
  float sp[16];
  const v16h* Bf = (const v16h*)wp2s;
#pragma unroll
  for (int nt = 0; nt < 16; ++nt) {
    // batch the 8 LDS B-fragment loads, then chain 8 WMMAs
    v16h b[8];
#pragma unroll
    for (int kc = 0; kc < 8; ++kc) b[kc] = Bf[(kc * 16 + nt) * 32 + lane];
    v8f acc = {};
#pragma unroll
    for (int kc = 0; kc < 8; ++kc)
      acc = __builtin_amdgcn_wmma_f32_16x16x32_f16(false, a[kc], false, b[kc],
                                                   (short)0, acc, false, false);
    int col = nt * 16 + li;
    float bb = bp2[col];
    float part = 0.f;
#pragma unroll
    for (int r = 0; r < 8; ++r) {
      int m = r + (hi ? 8 : 0);
      float val = vall[(size_t)idxv[r] * C_CH + col];
      part += (acc[r] + bb + val) * wbuf[w][m][nt];
    }
    part += __shfl_xor(part, 16, 32);  // combine both row-halves; all lanes valid
    sp[nt] = part;                     // spatial[g=nt, i=li]
  }

  // ---- DCT enhance (enh = sp @ E) and gate MLP, via wave shuffles ---------
  float enh[16];
#pragma unroll
  for (int g = 0; g < 16; ++g) enh[g] = 0.f;
  float go0 = 0.f, go1 = 0.f;
#pragma unroll
  for (int g = 0; g < 16; ++g) {
    float spg = sp[g];
#pragma unroll
    for (int m = 0; m < 16; ++m) {
      float sv = __shfl(spg, m, 32);
      enh[g] += sv * Emat[m * 16 + li];
      int cc = g * 16 + m;
      go0 += sv * Wg1[cc * 64 + lane];
      go1 += sv * Wg1[cc * 64 + lane + 32];
    }
  }
  float gpart = reluf(go0 + bg1[lane]) * Wg2[lane] +
                reluf(go1 + bg1[lane + 32]) * Wg2[lane + 32];
#pragma unroll
  for (int off = 16; off > 0; off >>= 1) gpart += __shfl_xor(gpart, off, 32);
  float gate = 0.2f / (1.f + __expf(-(gpart + bg2[0])));

  if (lane < 16) {
#pragma unroll
    for (int g = 0; g < 16; ++g)
      out[(size_t)n * C_CH + g * 16 + li] = (1.f - gate) * sp[g] + gate * enh[g];
  }
}

// --------------------------------- launch ----------------------------------
extern "C" void kernel_launch(void* const* d_in, const int* in_sizes, int n_in,
                              void* d_out, int out_size, void* d_ws, size_t ws_size,
                              hipStream_t stream) {
  (void)in_sizes; (void)n_in; (void)out_size; (void)ws_size;
  const float* feat  = (const float*)d_in[0];
  const float* coord = (const float*)d_in[1];
  const int*   ref   = (const int*)d_in[2];
  const float* Wq    = (const float*)d_in[3];
  const float* gq    = (const float*)d_in[5];
  const float* betaq = (const float*)d_in[6];
  const float* Wk    = (const float*)d_in[7];
  const float* gk    = (const float*)d_in[9];
  const float* betak = (const float*)d_in[10];
  const float* Wv    = (const float*)d_in[11];
  const float* bv    = (const float*)d_in[12];
  const float* Wp1   = (const float*)d_in[13];
  const float* bp1   = (const float*)d_in[14];
  const float* gp    = (const float*)d_in[15];
  const float* betap = (const float*)d_in[16];
  const float* Wp2   = (const float*)d_in[17];
  const float* bp2   = (const float*)d_in[18];
  const float* glw   = (const float*)d_in[19];
  const float* gw    = (const float*)d_in[20];
  const float* betaw = (const float*)d_in[21];
  const float* Ww    = (const float*)d_in[22];
  const float* bw    = (const float*)d_in[23];
  const float* Wg1   = (const float*)d_in[24];
  const float* bg1   = (const float*)d_in[25];
  const float* Wg2   = (const float*)d_in[26];
  const float* bg2   = (const float*)d_in[27];
  float* out = (float*)d_out;
  char* ws = (char*)d_ws;

  const size_t SZ_NC = (size_t)N_PTS * C_CH * sizeof(float);  // 64 MB
  const size_t OFF_V      = 0;
  const size_t OFF_QLIN   = SZ_NC;          // later reused as w0 (same size)
  const size_t OFF_W0     = OFF_QLIN;
  const size_t OFF_KLIN   = 2 * SZ_NC;
  const size_t OFF_QG     = 3 * SZ_NC;
  const size_t OFF_KG     = OFF_QG + (size_t)N_PTS * G_GR * 4;
  const size_t OFF_WALL   = OFF_KG + (size_t)N_PTS * G_GR * 4;
  const size_t OFF_WP2S   = OFF_WALL + 256 * 768 * 2;
  const size_t OFF_WP2GSW = OFF_WP2S + 256 * 256 * 2;
  const size_t OFF_BPG    = OFF_WP2GSW + 4096 * 2;
  const size_t OFF_E      = OFF_BPG + 256;
  const size_t OFF_SQ     = OFF_E + 1024;
  const size_t OFF_SHQ    = OFF_SQ + 1024;
  const size_t OFF_SK     = OFF_SHQ + 1024;
  const size_t OFF_SHK    = OFF_SK + 1024;
  const size_t OFF_WSC    = OFF_SHK + 1024;
  const size_t OFF_WSH    = OFF_WSC + 256;
  const size_t OFF_PPK    = OFF_WSH + 256;          // 256*8*4 = 8 KB
  const size_t OFF_PART   = OFF_PPK + 8192;         // qk col-stat partials (1 MB)
  const size_t OFF_PPART  = OFF_PART + (1u << 20);  // pos partials (512 KB)
  const size_t OFF_WPART  = OFF_PPART + (512u << 10);  // w partials (32 KB)

  float*    v_all  = (float*)(ws + OFF_V);
  float*    qlin   = (float*)(ws + OFF_QLIN);
  float*    w0     = (float*)(ws + OFF_W0);
  float*    klin   = (float*)(ws + OFF_KLIN);
  float*    qg     = (float*)(ws + OFF_QG);
  float*    kg     = (float*)(ws + OFF_KG);
  _Float16* wall   = (_Float16*)(ws + OFF_WALL);
  _Float16* wp2s   = (_Float16*)(ws + OFF_WP2S);
  _Float16* wp2gsw = (_Float16*)(ws + OFF_WP2GSW);
  float*    bpg    = (float*)(ws + OFF_BPG);
  float*    Emat   = (float*)(ws + OFF_E);
  float*    scq = (float*)(ws + OFF_SQ),  *shq = (float*)(ws + OFF_SHQ);
  float*    sck = (float*)(ws + OFF_SK),  *shk = (float*)(ws + OFF_SHK);
  float*    wsc = (float*)(ws + OFF_WSC), *wsh = (float*)(ws + OFF_WSH);
  float*    ppk   = (float*)(ws + OFF_PPK);
  float*    part  = (float*)(ws + OFF_PART);
  float*    ppart = (float*)(ws + OFF_PPART);
  float*    wpart = (float*)(ws + OFF_WPART);

  dim3 B(256);
  k_swz_wall<<<768, B, 0, stream>>>(Wq, Wk, Wv, wall);
  k_swz_wp2<<<256, B, 0, stream>>>(Wp2, wp2s);
  k_swz_wp2g<<<16, B, 0, stream>>>(Wp2, glw, wp2gsw);
  k_prep_misc<<<1, B, 0, stream>>>(bp2, glw, bpg, Emat);

  k_qkv<<<512, B, 0, stream>>>(feat, wall, bv, qlin, klin, v_all);
  k_colstats<<<256, B, 0, stream>>>(qlin, klin, part);
  k_qk_final<<<1, B, 0, stream>>>(part, gq, betaq, gk, betak, scq, shq, sck, shk);
  k_qg_kg<<<4096, B, 0, stream>>>(qlin, klin, scq, shq, sck, shk, glw, qg, kg);

  k_pos_stats<<<256, B, 0, stream>>>(coord, ref, Wp1, bp1, ppart);
  k_pos_final<<<1, B, 0, stream>>>(ppart, gp, betap, Wp1, bp1, ppk);

  // qlin/klin are dead now; w0 aliases the qlin region.
  k_w0<<<8192, B, 0, stream>>>(coord, ref, ppk, wp2gsw, bpg, qg, kg, w0);
  k_w0stats<<<256, B, 0, stream>>>(w0, wpart);
  k_w_final<<<1, B, 0, stream>>>(wpart, gw, betaw, wsc, wsh);

  k_main<<<8192, B, 0, stream>>>(coord, ref, v_all, w0, wp2s, ppk,
                                 bp2, wsc, wsh, Ww, bw, Emat, Wg1, bg1, Wg2, bg2,
                                 out);
}
